// SSIMLoss_80358838108879
// MI455X (gfx1250) — matrix operand order repaired and checked by reading, not compiled, not run on hardware
//
#include <hip/hip_runtime.h>

// ---------------------------------------------------------------------------
// Fused SSIM loss for MI455X (gfx1250, wave32).
//
// Roofline: 67 MB mandatory input @ 23.3 TB/s ~= 2.9 us -> single fused pass.
// Separable 11-tap Gaussian blur of the five SSIM fields as banded
// 16x16x32 f16 WMMAs (one WMMA = all 11 taps of one pass for a 16x16 tile).
// Each wave owns a 16x64 output column (4 stacked tiles). sigmoid(pred) and
// target are staged in LDS as vertical row-pair-packed f16 planes so WMMA
// B-operands assemble from raw b32 LDS loads with zero unpack moves; the
// vertical->horizontal transpose stages f32 and repacks to f16 A-layout with
// one v_cvt_pk_rtz per element pair. 67 MB working set is L2-resident
// (192 MB), so halo re-reads are L2 hits.
// ---------------------------------------------------------------------------

typedef __attribute__((ext_vector_type(16))) _Float16 v16h;
typedef __attribute__((ext_vector_type(2)))  _Float16 v2h;
typedef __attribute__((ext_vector_type(8)))  float    v8f;

#define IMG   512
#define NBAT  32
#define WAVES 4                  // waves per block
#define RTIL  4                  // stacked 16x16 tiles per wave (16x64 column)
#define SRP   40                 // staged row-pairs (80 rows incl. halo/pad)
#define PSTR  36                 // p/t plane row-pair stride (u32 words)
#define VSTR  34                 // transpose buffer row stride (f32 words)
#define NGRP  (NBAT * 32 * 8)    // 8192 wave-groups (32 tx * 8 ty-groups)
#define NBLK  (NGRP / WAVES)     // 2048 blocks
#define C1F   1.0e-4f            // 0.01^2
#define C2F   9.0e-4f            // 0.03^2

__global__ __launch_bounds__(WAVES * 32) void ssim_tile_kernel(
    const float* __restrict__ pred, const float* __restrict__ target,
    float* __restrict__ partial) {
  const int tid  = threadIdx.x;
  const int lane = tid & 31;
  const int wave = tid >> 5;
  const int half = lane >> 4;    // 16-lane half of the wave32
  const int l15  = lane & 15;

  __shared__ unsigned lds_p[WAVES][SRP * PSTR];  // {p[2r][c], p[2r+1][c]} pairs
  __shared__ unsigned lds_t[WAVES][SRP * PSTR];  // {t[2r][c], t[2r+1][c]} pairs
  __shared__ float    lds_v[WAVES][16 * VSTR];   // vertical blur, f32
  __shared__ float    red[WAVES];
  unsigned* pv = lds_p[wave];
  unsigned* tv = lds_t[wave];
  float*    vb = lds_v[wave];

  auto sig = [](float v) {
    return __builtin_amdgcn_rcpf(1.f + __expf(-v));
  };
  auto pack2 = [](float a, float b) {
    return __builtin_bit_cast(unsigned, __builtin_amdgcn_cvt_pkrtz(a, b));
  };
  auto pack2h = [](float a, float b) {
    return __builtin_bit_cast(v2h, __builtin_amdgcn_cvt_pkrtz(a, b));
  };

  // Normalization of the 1D Gaussian window (sigma=1.5, ws=11).
  float inv;
  {
    float s = 0.f;
#pragma unroll
    for (int i = 0; i < 11; ++i) {
      float x = (float)(i - 5);
      s += __expf(x * x * (-1.f / 4.5f));
    }
    inv = __builtin_amdgcn_rcpf(s);
  }

  // Banded weight operands, computed analytically. Layouts used here (contiguous K, same for A and B):
  //   A (16x32 f16): M = lane&15; elem e -> K = e + 16*half
  //   B (32x16 f16): N = lane&15; elem e -> K = e + 16*half
  v16h wv, wh;
#pragma unroll
  for (int e = 0; e < 16; ++e) {
    const int kA = e + 16 * half;
    const int dA = kA - l15;                        // Wv[m][k] = w[k-m]
    const float xa = (float)(dA - 5);
    const float va = __expf(xa * xa * (-1.f / 4.5f)) * inv;
    wv[e] = ((unsigned)dA <= 10u) ? (_Float16)va : (_Float16)0.f;
    const int kB = e + 16 * half;
    const int dB = kB - l15;                        // Wh[k][n] = w[k-n]
    const float xb = (float)(dB - 5);
    const float vbw = __expf(xb * xb * (-1.f / 4.5f)) * inv;
    wh[e] = ((unsigned)dB <= 10u) ? (_Float16)vbw : (_Float16)0.f;
  }

  // Wave-group -> (image n, 16x64 output column at x0, y0).
  const int g   = blockIdx.x * WAVES + wave;
  const int n   = g >> 8;
  const int rem = g & 255;
  const int y0  = (rem >> 5) << 6;
  const int x0  = (rem & 31) << 4;
  const float* pbase = pred   + (size_t)n * (IMG * IMG);
  const float* tbase = target + (size_t)n * (IMG * IMG);

  // Stage 80x32 halo'd region as row-pair-packed f16 planes.
  const bool interior = (x0 >= 16) && (x0 <= 480) && (y0 >= 64) && (y0 <= 384);
  if (interior) {
    // Unchecked path: float4 (b128) global loads; each lane covers one
    // row-pair x 4 cols; one uint4 (b128) LDS store per plane.
    const int sub = lane & 7;
    const int rpo = lane >> 3;
    const int xc  = x0 - 5 + sub * 4;
#pragma unroll 2
    for (int it = 0; it < SRP / 4; ++it) {
      const int rp = it * 4 + rpo;
      const int y  = y0 - 5 + 2 * rp;
      const float4 pa = *reinterpret_cast<const float4*>(pbase + y * IMG + xc);
      const float4 pb = *reinterpret_cast<const float4*>(pbase + (y + 1) * IMG + xc);
      const float4 ta = *reinterpret_cast<const float4*>(tbase + y * IMG + xc);
      const float4 tb = *reinterpret_cast<const float4*>(tbase + (y + 1) * IMG + xc);
      uint4 wp, wt;
      wp.x = pack2(sig(pa.x), sig(pb.x));
      wp.y = pack2(sig(pa.y), sig(pb.y));
      wp.z = pack2(sig(pa.z), sig(pb.z));
      wp.w = pack2(sig(pa.w), sig(pb.w));
      wt.x = pack2(ta.x, tb.x);
      wt.y = pack2(ta.y, tb.y);
      wt.z = pack2(ta.z, tb.z);
      wt.w = pack2(ta.w, tb.w);
      *reinterpret_cast<uint4*>(pv + rp * PSTR + sub * 4) = wp;
      *reinterpret_cast<uint4*>(tv + rp * PSTR + sub * 4) = wt;
    }
  } else {
    // Border path: scalar loads, zero-padded convolution boundary.
    const int x = x0 - 5 + lane;
    const bool xin = (x >= 0) && (x < IMG);
    for (int rp = 0; rp < SRP; ++rp) {
      const int ya = y0 - 5 + 2 * rp;
      float p0 = 0.f, p1 = 0.f, t0 = 0.f, t1 = 0.f;
      if (xin) {
        if (ya >= 0 && ya < IMG) {
          p0 = sig(pbase[ya * IMG + x]);
          t0 = tbase[ya * IMG + x];
        }
        if (ya + 1 >= 0 && ya + 1 < IMG) {
          p1 = sig(pbase[(ya + 1) * IMG + x]);
          t1 = tbase[(ya + 1) * IMG + x];
        }
      }
      pv[rp * PSTR + lane] = pack2(p0, p1);
      tv[rp * PSTR + lane] = pack2(t0, t1);
    }
  }

  const v8f zero8 = {};
  // Separable blur of one 16x16 tile: vertical L/R WMMAs, f32 LDS transpose,
  // repack to f16 A-layout element pairs, horizontal WMMA.
  auto blur = [&](v16h bl, v16h br) -> v8f {
    v8f vl = __builtin_amdgcn_wmma_f32_16x16x32_f16(
        false, wv, false, bl, (short)0, zero8, false, false);
    v8f vr = __builtin_amdgcn_wmma_f32_16x16x32_f16(
        false, wv, false, br, (short)0, zero8, false, false);
#pragma unroll
    for (int d = 0; d < 8; ++d) {                  // D elem d: M=d+8*half, N=l15
      const int m = d + 8 * half;
      vb[m * VSTR + l15]      = vl[d];
      vb[m * VSTR + 16 + l15] = vr[d];
    }
    v16h ah;                                       // A elem pair (2j,2j+1):
#pragma unroll                                     //  K = 2j + 16*half
    for (int j = 0; j < 8; ++j) {
      const int koff = 2 * j + 16 * half;
      const float2 u = *reinterpret_cast<const float2*>(vb + l15 * VSTR + koff);
      const v2h pr = pack2h(u.x, u.y);
      ah[2 * j]     = pr[0];
      ah[2 * j + 1] = pr[1];
    }
    return __builtin_amdgcn_wmma_f32_16x16x32_f16(
        false, ah, false, wh, (short)0, zero8, false, false);
  };

  float acc = 0.f;
#pragma unroll 1
  for (int b = 0; b < RTIL; ++b) {
    // B-operands: elem pair (2j,2j+1) = rows 16h+2j, 16h+2j+1 of the staged
    // block = one packed u32 per plane -> direct register placement.
    v16h pL, tL, pR, tR;
#pragma unroll
    for (int j = 0; j < 8; ++j) {
      const int rp = 8 * b + 8 * half + j;
      const v2h a = __builtin_bit_cast(v2h, pv[rp * PSTR + l15]);
      const v2h bb = __builtin_bit_cast(v2h, tv[rp * PSTR + l15]);
      const v2h c = __builtin_bit_cast(v2h, pv[rp * PSTR + 16 + l15]);
      const v2h dd = __builtin_bit_cast(v2h, tv[rp * PSTR + 16 + l15]);
      pL[2 * j] = a[0];  pL[2 * j + 1] = a[1];
      tL[2 * j] = bb[0]; tL[2 * j + 1] = bb[1];
      pR[2 * j] = c[0];  pR[2 * j + 1] = c[1];
      tR[2 * j] = dd[0]; tR[2 * j + 1] = dd[1];
    }
    v8f Hm1 = blur(pL, pR);
    v8f Hm2 = blur(tL, tR);
    v8f Hpp = blur(pL * pL, pR * pR);              // vector v_pk_mul_f16
    v8f Htt = blur(tL * tL, tR * tR);
    v8f Hpt = blur(pL * tL, pR * tR);

#pragma unroll
    for (int d = 0; d < 8; ++d) {
      const float mu1 = Hm1[d], mu2 = Hm2[d];
      const float mu1s = mu1 * mu1, mu2s = mu2 * mu2, m12 = mu1 * mu2;
      const float s12 = Hpt[d] - m12;
      const float num = (2.f * m12 + C1F) * (2.f * s12 + C2F);
      const float den = (mu1s + mu2s + C1F) *
                        ((Hpp[d] - mu1s) + (Htt[d] - mu2s) + C2F);
      acc += num * __builtin_amdgcn_rcpf(den);
    }
  }

  // Deterministic in-block reduction.
#pragma unroll
  for (int off = 16; off > 0; off >>= 1) acc += __shfl_xor(acc, off, 32);
  if (lane == 0) red[wave] = acc;
  __syncthreads();
  if (tid == 0)
    partial[blockIdx.x] = (red[0] + red[1]) + (red[2] + red[3]);
}

// Fixed-order single-block reduction => bit-deterministic across replays.
__global__ __launch_bounds__(256) void ssim_reduce_kernel(
    const float* __restrict__ partial, float* __restrict__ out) {
  __shared__ float s[256];
  const int tid = threadIdx.x;
  float a = 0.f;
  for (int i = tid; i < NBLK; i += 256) a += partial[i];
  s[tid] = a;
  __syncthreads();
  for (int off = 128; off > 0; off >>= 1) {
    if (tid < off) s[tid] += s[tid + off];
    __syncthreads();
  }
  if (tid == 0)
    out[0] = 1.f - s[0] * (1.f / (float)(NBAT * IMG * IMG));
}

extern "C" void kernel_launch(void* const* d_in, const int* in_sizes, int n_in,
                              void* d_out, int out_size, void* d_ws, size_t ws_size,
                              hipStream_t stream) {
  const float* pred   = (const float*)d_in[0];
  const float* target = (const float*)d_in[1];
  float* out     = (float*)d_out;
  float* partial = (float*)d_ws;  // NBLK floats = 8 KB scratch

  ssim_tile_kernel<<<NBLK, WAVES * 32, 0, stream>>>(pred, target, partial);
  ssim_reduce_kernel<<<1, 256, 0, stream>>>(partial, out);
}
